// InferenceEngine_87316685128498
// MI455X (gfx1250) — compile-verified, matrix operand order checked by reading, not run on hardware
//
#include <hip/hip_runtime.h>
#include <hip/hip_bf16.h>

typedef __attribute__((ext_vector_type(16))) __bf16 v16bf;
typedef __attribute__((ext_vector_type(8)))  float  v8f;

// Types for the gfx1250 async global->LDS builtin:
// param0: v4i in address_space(1) (global), param1: v4i in address_space(3) (LDS)
typedef int v4i __attribute__((vector_size(4 * sizeof(int))));
typedef __attribute__((address_space(1))) v4i as1_v4i;
typedef __attribute__((address_space(3))) v4i as3_v4i;

#define B_    256
#define D_    49152
#define DM_   512
#define NDOM  5
#define E_    6
#define H1_   256
#define C_    10
#define KS_   8                  // K-splits for backbone GEMM
#define KCHUNK_ (D_ / KS_)       // 6144

// Output layout (floats): logits[256*10] | gates[256*5] | entropy[256] | eid[256] | is_ood[256]
#define O_GATES (B_ * C_)
#define O_ENT   (O_GATES + B_ * NDOM)
#define O_EID   (O_ENT + B_)
#define O_OOD   (O_EID + B_)

// CDNA5 async global->LDS path, guarded so both host and device passes compile
#if defined(__AMDGCN__) && __has_builtin(__builtin_amdgcn_global_load_async_to_lds_b128) && __has_builtin(__builtin_amdgcn_s_wait_asynccnt)
#define USE_ASYNC_LDS 1
#else
#define USE_ASYNC_LDS 0
#endif

__device__ __forceinline__ unsigned int pk_bf16(float lo, float hi) {
    // lowers to v_cvt_pk_bf16_f32
    union { __bf16 h[2]; unsigned int u; } v;
    v.h[0] = (__bf16)lo; v.h[1] = (__bf16)hi;
    return v.u;
}

__device__ __forceinline__ v16bf cvt_a_frag(const float* __restrict__ p) {
    // 16-bit A-matrix 16x32 layout: lane<16 holds K = koff+0..7 and koff+16..23 (koff folded by caller)
    float4 q0 = *(const float4*)(p);
    float4 q1 = *(const float4*)(p + 4);
    float4 q2 = *(const float4*)(p + 16);
    float4 q3 = *(const float4*)(p + 20);
    v16bf a;
    a[0]  = (__bf16)q0.x; a[1]  = (__bf16)q0.y; a[2]  = (__bf16)q0.z; a[3]  = (__bf16)q0.w;
    a[4]  = (__bf16)q1.x; a[5]  = (__bf16)q1.y; a[6]  = (__bf16)q1.z; a[7]  = (__bf16)q1.w;
    a[8]  = (__bf16)q2.x; a[9]  = (__bf16)q2.y; a[10] = (__bf16)q2.z; a[11] = (__bf16)q2.w;
    a[12] = (__bf16)q3.x; a[13] = (__bf16)q3.y; a[14] = (__bf16)q3.z; a[15] = (__bf16)q3.w;
    return a;
}

// ---------------------------------------------------------------------------
// Kernel 1: backbone GEMM partials. zpart[ks][m][n] = sum over K-chunk of x*Wb
// Grid: 64 blocks (ks in [0,8), mt in [0,2), nt4 in [0,4)), 256 threads (8 waves).
// WG tile: 128(M) x 128(N); per-wave 16 rows x 8 n-tiles.
// B tile is staged in LDS as bf16 in fragment-ready [n][k] layout (stride 40
// ushorts = 80B, 16B aligned) so every B fragment is exactly 2x ds_load_b128
// and the f32->bf16 conversion happens once per element (not once per wave).
// ---------------------------------------------------------------------------
__global__ __launch_bounds__(256)
void backbone_gemm(const float* __restrict__ x, const float* __restrict__ Wb,
                   float* __restrict__ zpart) {
    __shared__ __align__(16) unsigned short btile16[128 * 40];

    const int bx    = blockIdx.x;
    const int ks    = bx & 7;
    const int mt    = (bx >> 3) & 1;
    const int nt4   = bx >> 4;
    const int mbase = mt * 128;
    const int nbase = nt4 * 128;
    const int kb0   = ks * KCHUNK_;

    const int tid   = threadIdx.x;
    const int lane  = tid & 31;
    const int wave  = tid >> 5;
    const int mlane = lane & 15;
    const int koff  = (lane < 16) ? 0 : 8;

    const int sn    = tid & 127;   // staging column 0..127
    const int shalf = tid >> 7;    // 0..1

    const float* xrow = x + (size_t)(mbase + wave * 16 + mlane) * D_;

    v8f acc[8];
#pragma unroll
    for (int i = 0; i < 8; ++i) acc[i] = (v8f){0.f,0.f,0.f,0.f,0.f,0.f,0.f,0.f};

    for (int kk = 0; kk < KCHUNK_; kk += 32) {
        const int kb = kb0 + kk;
        __syncthreads();   // previous iteration's LDS reads complete
        // stage 32(K) x 128(N) tile, converting to bf16, K-contiguous per column
#pragma unroll
        for (int j = 0; j < 2; ++j) {
            const int k0 = shalf * 8 + j * 16;        // {0,16} or {8,24}
            const float* src = Wb + (size_t)(kb + k0) * DM_ + nbase + sn;
            float v0 = src[0 * DM_], v1 = src[1 * DM_];
            float v2 = src[2 * DM_], v3 = src[3 * DM_];
            float v4 = src[4 * DM_], v5 = src[5 * DM_];
            float v6 = src[6 * DM_], v7 = src[7 * DM_];
            uint4 pk;
            pk.x = pk_bf16(v0, v1); pk.y = pk_bf16(v2, v3);
            pk.z = pk_bf16(v4, v5); pk.w = pk_bf16(v6, v7);
            *(uint4*)(&btile16[sn * 40 + k0]) = pk;
        }
        __syncthreads();

        __builtin_prefetch(xrow + kb + 64, 0, 0);   // global_prefetch_b8 on the A stream

        v16bf a = cvt_a_frag(xrow + kb + koff);

#pragma unroll
        for (int nt = 0; nt < 8; ++nt) {
            const int cn = nt * 16 + mlane;
            union { v16bf v; uint4 q[2]; } bf;
            bf.q[0] = *(const uint4*)(&btile16[cn * 40 + koff]);       // K=koff..koff+7
            bf.q[1] = *(const uint4*)(&btile16[cn * 40 + koff + 16]);  // K=koff+16..koff+23
            acc[nt] = __builtin_amdgcn_wmma_f32_16x16x32_bf16(
                false, a, false, bf.v, (short)0, acc[nt], false, false);
        }
    }

    // write partials: C layout -> VGPR r: lanes 0-15 M=r, lanes 16-31 M=r+8, N=lane&15
    float* zp = zpart + (size_t)ks * B_ * DM_;
#pragma unroll
    for (int nt = 0; nt < 8; ++nt) {
        const int nn = nbase + nt * 16 + mlane;
#pragma unroll
        for (int r = 0; r < 8; ++r) {
            const int mm = mbase + wave * 16 + r + ((lane < 16) ? 0 : 8);
            zp[(size_t)mm * DM_ + nn] = acc[nt][r];
        }
    }
}

// ---------------------------------------------------------------------------
// Kernel 2: reduce split-K partials + bias + ReLU -> z
// ---------------------------------------------------------------------------
__global__ __launch_bounds__(256)
void bias_relu_reduce(const float* __restrict__ zpart, const float* __restrict__ bb,
                      float* __restrict__ z) {
    int i = blockIdx.x * 256 + threadIdx.x;
    if (i >= B_ * DM_) return;
    float s = 0.f;
#pragma unroll
    for (int k = 0; k < KS_; ++k) s += zpart[(size_t)k * B_ * DM_ + i];
    s += bb[i % DM_];
    z[i] = s > 0.f ? s : 0.f;
}

// ---------------------------------------------------------------------------
// Kernel 3: router. One wave32 per sample: 5 dot products of 512, softmax,
// entropy, argmax, OOD test, expert dispatch lists via int atomics.
// ---------------------------------------------------------------------------
__global__ __launch_bounds__(32)
void router(const float* __restrict__ z, const float* __restrict__ Wg,
            const float* __restrict__ bg, float* __restrict__ out,
            int* __restrict__ counts, int* __restrict__ lists) {
    const int b = blockIdx.x;
    const int lane = threadIdx.x;

    float p[NDOM] = {0.f, 0.f, 0.f, 0.f, 0.f};
    for (int k = lane; k < DM_; k += 32) {
        float zv = z[b * DM_ + k];
#pragma unroll
        for (int j = 0; j < NDOM; ++j) p[j] += zv * Wg[k * NDOM + j];
    }
#pragma unroll
    for (int off = 16; off > 0; off >>= 1) {
#pragma unroll
        for (int j = 0; j < NDOM; ++j) p[j] += __shfl_down(p[j], off, 32);
    }

    if (lane == 0) {
#pragma unroll
        for (int j = 0; j < NDOM; ++j) p[j] += bg[j];
        float mx = p[0];
#pragma unroll
        for (int j = 1; j < NDOM; ++j) mx = p[j] > mx ? p[j] : mx;
        float ex[NDOM], sum = 0.f;
#pragma unroll
        for (int j = 0; j < NDOM; ++j) { ex[j] = expf(p[j] - mx); sum += ex[j]; }
        float inv = 1.f / sum;
        float ent = 0.f;
        int amax = 0;
#pragma unroll
        for (int j = 0; j < NDOM; ++j) {
            float g = ex[j] * inv;
            ex[j] = g;
            ent -= g * logf(g + 1e-8f);
            if (g > ex[amax]) amax = j;   // first-max kept (strict >)
        }
        const bool ood = ent > 0.80471895621705016f;  // log(5)/2
        const int eid = ood ? 5 : amax;
#pragma unroll
        for (int j = 0; j < NDOM; ++j) out[O_GATES + b * NDOM + j] = ex[j];
        out[O_ENT + b] = ent;
        out[O_EID + b] = (float)eid;
        out[O_OOD + b] = ood ? 1.f : 0.f;
        int pos = atomicAdd(&counts[eid], 1);
        lists[eid * B_ + pos] = b;
    }
}

// ---------------------------------------------------------------------------
// Kernel 4: sparse expert dispatch. Grid = 6 experts x 16 row-tiles.
// Each active block gathers <=16 sample rows, computes
// h = relu(Xg[16,49152] @ W1[e][49152,256] + b1[e]) via bf16 WMMA,
// then logits = h @ W2[e] + b2[e] through LDS.
// W1 tile streamed into LDS with CDNA5 async global->LDS copies (ASYNCcnt);
// each column group is owned by exactly one wave, so the per-wave f32->bf16
// conversion here is not redundant.
// ---------------------------------------------------------------------------
__global__ __launch_bounds__(256)
void experts(const float* __restrict__ x, const float* __restrict__ W1,
             const float* __restrict__ b1, const float* __restrict__ W2,
             const float* __restrict__ b2, const int* __restrict__ counts,
             const int* __restrict__ lists, float* __restrict__ out) {
    __shared__ float wtile[32 * 260];   // 32 K-rows x 256 cols, padded stride 260
    __shared__ float hbuf[16 * 256];
    __shared__ int   rows[16];

    const int e = blockIdx.x >> 4;
    const int t = blockIdx.x & 15;
    const int count = counts[e];
    if (t * 16 >= count) return;        // inactive tile: no WMMA issued after this
    int nrows = count - t * 16;
    if (nrows > 16) nrows = 16;

    const int tid   = threadIdx.x;
    const int lane  = tid & 31;
    const int wave  = tid >> 5;
    const int mlane = lane & 15;
    const int koff  = (lane < 16) ? 0 : 8;

    if (tid < 16) rows[tid] = lists[e * B_ + t * 16 + (tid < nrows ? tid : 0)];
    __syncthreads();

    const float* xrow = x + (size_t)rows[mlane] * D_;
    const float* W1e  = W1 + (size_t)e * D_ * H1_;

    v8f acc0 = (v8f){0.f,0.f,0.f,0.f,0.f,0.f,0.f,0.f};
    v8f acc1 = (v8f){0.f,0.f,0.f,0.f,0.f,0.f,0.f,0.f};

    for (int kb = 0; kb < D_; kb += 32) {
        __syncthreads();
#if USE_ASYNC_LDS
#pragma unroll
        for (int j = 0; j < 8; ++j) {
            int f4 = tid + 256 * j;            // 2048 float4 transfers
            int r  = f4 >> 6;
            int c4 = f4 & 63;
            __builtin_amdgcn_global_load_async_to_lds_b128(
                (as1_v4i*)(W1e + (size_t)(kb + r) * H1_ + c4 * 4),
                (as3_v4i*)(&wtile[r * 260 + c4 * 4]), 0, 0);
        }
        __builtin_amdgcn_s_wait_asynccnt(0);
#else
#pragma unroll
        for (int j = 0; j < 8; ++j) {
            int f4 = tid + 256 * j;
            int r  = f4 >> 6;
            int c4 = f4 & 63;
            float4 v = *(const float4*)(W1e + (size_t)(kb + r) * H1_ + c4 * 4);
            *(float4*)(&wtile[r * 260 + c4 * 4]) = v;
        }
#endif
        __syncthreads();

        __builtin_prefetch(xrow + kb + 64, 0, 0);

        v16bf a = cvt_a_frag(xrow + kb + koff);

#pragma unroll
        for (int nt = 0; nt < 2; ++nt) {
            const int cn = wave * 32 + nt * 16 + mlane;
            v16bf b;
#pragma unroll
            for (int i = 0; i < 8; ++i) {
                b[i]     = (__bf16)wtile[(koff + i) * 260 + cn];
                b[8 + i] = (__bf16)wtile[(koff + 16 + i) * 260 + cn];
            }
            if (nt == 0)
                acc0 = __builtin_amdgcn_wmma_f32_16x16x32_bf16(
                    false, a, false, b, (short)0, acc0, false, false);
            else
                acc1 = __builtin_amdgcn_wmma_f32_16x16x32_bf16(
                    false, a, false, b, (short)0, acc1, false, false);
        }
    }

    // epilogue: bias + relu -> hbuf
#pragma unroll
    for (int nt = 0; nt < 2; ++nt) {
        const int col = wave * 32 + nt * 16 + mlane;
        const float bias = b1[e * H1_ + col];
#pragma unroll
        for (int r = 0; r < 8; ++r) {
            const int mm = r + ((lane < 16) ? 0 : 8);
            float v = (nt == 0 ? acc0[r] : acc1[r]) + bias;
            hbuf[mm * 256 + col] = v > 0.f ? v : 0.f;
        }
    }
    __syncthreads();

    // logits = h @ W2[e] + b2[e]; 160 outputs, 256-deep reduction each
    if (tid < 16 * C_) {
        const int row = tid / C_;
        const int c   = tid % C_;
        float s = 0.f;
        for (int k = 0; k < H1_; ++k)
            s += hbuf[row * 256 + k] * W2[((size_t)e * H1_ + k) * C_ + c];
        s += b2[e * C_ + c];
        if (row < nrows) out[rows[row] * C_ + c] = s;
    }
}

// ---------------------------------------------------------------------------
extern "C" void kernel_launch(void* const* d_in, const int* in_sizes, int n_in,
                              void* d_out, int out_size, void* d_ws, size_t ws_size,
                              hipStream_t stream) {
    const float* x  = (const float*)d_in[0];
    const float* Wb = (const float*)d_in[1];
    const float* bb = (const float*)d_in[2];
    const float* Wg = (const float*)d_in[3];
    const float* bg = (const float*)d_in[4];
    const float* W1 = (const float*)d_in[5];
    const float* b1 = (const float*)d_in[6];
    const float* W2 = (const float*)d_in[7];
    const float* b2 = (const float*)d_in[8];
    float* out = (float*)d_out;

    // ws layout (floats): zpart[KS*B*DM] | z[B*DM] | counts[16 ints] | lists[E*B ints]
    float* zpart  = (float*)d_ws;
    float* z      = zpart + (size_t)KS_ * B_ * DM_;
    int*   counts = (int*)(z + (size_t)B_ * DM_);
    int*   lists  = counts + 16;

    (void)hipMemsetAsync(counts, 0, 16 * sizeof(int), stream);
    backbone_gemm<<<dim3(64), dim3(256), 0, stream>>>(x, Wb, zpart);
    bias_relu_reduce<<<dim3((B_ * DM_ + 255) / 256), dim3(256), 0, stream>>>(zpart, bb, z);
    router<<<dim3(B_), dim3(32), 0, stream>>>(z, Wg, bg, out, counts, lists);
    experts<<<dim3(E_ * 16), dim3(256), 0, stream>>>(x, W1, b1, W2, b2, counts, lists, out);
}
